// Attention_21303037788751
// MI455X (gfx1250) — matrix operand order repaired and checked by reading, not compiled, not run on hardware
//
#include <hip/hip_runtime.h>
#include <math.h>

// ---- CDNA5 WMMA vector types --------------------------------------------
typedef __attribute__((ext_vector_type(16))) __bf16 v16bf;
typedef __attribute__((ext_vector_type(8)))  __bf16 v8bf;
typedef __attribute__((ext_vector_type(8)))  float  v8f;
typedef __attribute__((ext_vector_type(4)))  int    v4i;

#define BATCH 16
#define SEQ   512
#define FDIM  256
#define NHEAD 8
#define UNITS 64
#define INNER (NHEAD*UNITS)   // 512
#define QKVC  (3*INNER)       // 1536

// ---- feature detection for gfx1250 async-to-LDS --------------------------
#if defined(__has_builtin)
#if __has_builtin(__builtin_amdgcn_global_load_async_to_lds_b128)
#define HAVE_ASYNC_LDS 1
#endif
#if __has_builtin(__builtin_amdgcn_s_wait_asynccnt)
#define HAVE_WAIT_ASYNC_BUILTIN 1
#endif
#endif

typedef __attribute__((address_space(1))) v4i g1_v4i;   // global 128-bit chunk
typedef __attribute__((address_space(3))) v4i l3_v4i;   // LDS 128-bit chunk

// 16-byte global -> LDS copy (async on CDNA5, tracked by ASYNCcnt)
__device__ __forceinline__ void cp16_to_lds(__bf16* dst_lds, const __bf16* src) {
#ifdef HAVE_ASYNC_LDS
  __builtin_amdgcn_global_load_async_to_lds_b128(
      (g1_v4i*)src, (l3_v4i*)dst_lds, 0, 0);
#else
  *(v8bf*)dst_lds = *(const v8bf*)src;
#endif
}

__device__ __forceinline__ void wait_async_lds() {
#ifdef HAVE_ASYNC_LDS
#ifdef HAVE_WAIT_ASYNC_BUILTIN
  __builtin_amdgcn_s_wait_asynccnt(0);
#else
  asm volatile("s_wait_asynccnt 0" ::: "memory");
#endif
#endif
}

// f32 -> bf16, round-to-nearest-even
__device__ __forceinline__ __bf16 f2bf(float f) {
  unsigned u = __builtin_bit_cast(unsigned, f);
  u += 0x7fffu + ((u >> 16) & 1u);
  unsigned short h = (unsigned short)(u >> 16);
  return __builtin_bit_cast(__bf16, h);
}

// A fragment (16x32 bf16, row-major, leading dim ld).
// Lane l holds row m=l&15; lanes 0-15: K=0..15, lanes 16-31: K=16..31.
__device__ __forceinline__ v16bf load_a(const __bf16* A, int ld, int lane) {
  int m  = lane & 15;
  int kb = (lane >> 4) << 4;                    // 0 or 16
  const __bf16* p = A + (size_t)m * ld + kb;
  v8bf lo = *(const v8bf*)(p);
  v8bf hi = *(const v8bf*)(p + 8);
  return __builtin_shufflevector(lo, hi, 0,1,2,3,4,5,6,7,8,9,10,11,12,13,14,15);
}

// B fragment (32x16 bf16). Bt column-major (column contiguous), leading dim ld.
// Lane l holds column n=l&15; lanes 0-15: K=0..15, lanes 16-31: K=16..31.
__device__ __forceinline__ v16bf load_b(const __bf16* Bt, int ld, int lane) {
  int n  = lane & 15;
  int kb = (lane >> 4) << 4;                    // 0 or 16
  return *(const v16bf*)(Bt + (size_t)n * ld + kb);
}

__device__ __forceinline__ v8f wmma_bf(v16bf a, v16bf b, v8f c) {
  return __builtin_amdgcn_wmma_f32_16x16x32_bf16(false, a, false, b,
                                                 (short)0, c, false, false);
}

// ---- Stage 0: conversions -----------------------------------------------
__global__ void cvt_f32_bf16(const float* __restrict__ x, __bf16* __restrict__ y, int n) {
  int i = blockIdx.x * 256 + threadIdx.x;
  if (i < n) y[i] = f2bf(x[i]);
}

// w: rows x cols row-major -> wt: cols x rows (column-major for WMMA B)
__global__ void transpose_w(const float* __restrict__ w, __bf16* __restrict__ wt,
                            int rows, int cols) {
  int i = blockIdx.x * 256 + threadIdx.x;
  if (i < rows * cols) {
    int r = i / cols, c = i % cols;
    wt[(size_t)c * rows + r] = f2bf(w[i]);
  }
}

// ---- Stage 1: QKV GEMM ---------------------------------------------------
// X: (8192 x 256) bf16, Wt: (1536 cols x 256) col-major bf16.
// Block: 4 waves share an LDS-staged A row tile; each wave does 2 col tiles.
__device__ __forceinline__ void qkv_scatter(const v8f& c, int n0, int rt, int lane,
                                            __bf16* q, __bf16* k, __bf16* vT) {
  int h   = n0 / (3 * UNITS);
  int rem = n0 % (3 * UNITS);
  int s   = rem / UNITS;
  int u   = rem % UNITS + (lane & 15);
  int b_  = rt >> 5;
  int i0  = (rt & 31) * 16;
  int mh  = (lane >> 4) << 3;
  size_t bh = (size_t)b_ * NHEAD + h;
  if (s == 0) {
    for (int r = 0; r < 8; ++r)
      q[(bh * SEQ + i0 + mh + r) * UNITS + u] = f2bf(c[r]);
  } else if (s == 1) {
    for (int r = 0; r < 8; ++r)
      k[(bh * SEQ + i0 + mh + r) * UNITS + u] = f2bf(c[r]);
  } else {
    for (int r = 0; r < 8; ++r)
      vT[(bh * UNITS + u) * SEQ + i0 + mh + r] = f2bf(c[r]);
  }
}

__global__ void qkv_gemm(const __bf16* __restrict__ X, const __bf16* __restrict__ Wt,
                         __bf16* __restrict__ q, __bf16* __restrict__ k,
                         __bf16* __restrict__ vT) {
  __shared__ __align__(32) __bf16 As[16 * FDIM];         // 8 KB shared A tile
  int lane = threadIdx.x & 31;
  int wave = threadIdx.x >> 5;
  int rt = blockIdx.x;                   // 512 row tiles of 16 over 8192 rows
  int n0 = (blockIdx.y * 4 + wave) * 32; // wave covers 2 col tiles (32 cols)

  // cooperative async stage of the 16x256 A tile
  const __bf16* Arow = X + (size_t)rt * 16 * FDIM;
  for (int t = threadIdx.x; t < (16 * FDIM) / 8; t += blockDim.x)
    cp16_to_lds(As + t * 8, Arow + t * 8);
  wait_async_lds();
  __syncthreads();

  const __bf16* Bt0 = Wt + (size_t)n0 * FDIM;
  const __bf16* Bt1 = Bt0 + (size_t)16 * FDIM;
  v8f c0 = {}, c1 = {};
  for (int kt = 0; kt < FDIM / 32; ++kt) {
    v16bf a  = load_a(As + kt * 32, FDIM, lane);
    v16bf b0 = load_b(Bt0 + kt * 32, FDIM, lane);
    v16bf b1 = load_b(Bt1 + kt * 32, FDIM, lane);
    c0 = wmma_bf(a, b0, c0);
    c1 = wmma_bf(a, b1, c1);
  }
  qkv_scatter(c0, n0,      rt, lane, q, k, vT);
  qkv_scatter(c1, n0 + 16, rt, lane, q, k, vT);
}

// ---- Stage 2: fused attention -------------------------------------------
// One block = 4 waves handles one (b,h, 16-row tile). Full row (N=512) of
// scores lives in LDS -> plain softmax, no online rescale.
__global__ void attn_fused(const __bf16* __restrict__ q, const __bf16* __restrict__ k,
                           const __bf16* __restrict__ vT,
                           const float* __restrict__ adj, const float* __restrict__ dist,
                           __bf16* __restrict__ O) {
  __shared__ __align__(32) float  Sf[16 * SEQ];    // 32 KB scores
  __shared__ __align__(32) __bf16 Abf[16 * SEQ];   // 16 KB attn (bf16)
  __shared__ float red[16 * 8];

  int lane = threadIdx.x & 31;
  int wave = threadIdx.x >> 5;
  int bh = blockIdx.y;                   // 0..127  (b*8+h)
  int b_ = bh >> 3;
  int it = blockIdx.x;                   // 0..31 row tile
  const float scale = 0.125f;            // 1/sqrt(UNITS)

  // prefetch the adj/dist rows this block will consume exactly once
  {
    size_t rowbase = ((size_t)b_ * SEQ + it * 16) * SEQ;
    for (int t = threadIdx.x; t < (16 * SEQ) / 32; t += blockDim.x) {
      __builtin_prefetch(adj  + rowbase + (size_t)t * 32, 0, 1);
      __builtin_prefetch(dist + rowbase + (size_t)t * 32, 0, 1);
    }
  }

  // ---- S = (q . k^T) * scale : each wave covers 8 j-tiles of 16
  const __bf16* qbase = q + ((size_t)bh * SEQ + it * 16) * UNITS;
  v16bf a0 = load_a(qbase,      UNITS, lane);
  v16bf a1 = load_a(qbase + 32, UNITS, lane);
  int n  = lane & 15;
  int mh = (lane >> 4) << 3;
  for (int j8 = 0; j8 < 8; ++j8) {
    int jt = wave * 8 + j8;
    const __bf16* kb = k + ((size_t)bh * SEQ + jt * 16) * UNITS;
    v8f c = {};
    c = wmma_bf(a0, load_b(kb,      UNITS, lane), c);
    c = wmma_bf(a1, load_b(kb + 32, UNITS, lane), c);
    for (int r = 0; r < 8; ++r)
      Sf[(mh + r) * SEQ + jt * 16 + n] = c[r] * scale;
  }
  __syncthreads();

  // ---- softmax + combine: thread t owns a 64-wide strip of row t>>3
  {
    int r  = threadIdx.x >> 3;
    int c0 = (threadIdx.x & 7) * 64;
    float m = -1e30f;
    for (int j = 0; j < 64; ++j) m = fmaxf(m, Sf[r * SEQ + c0 + j]);
    red[r * 8 + (threadIdx.x & 7)] = m;
    __syncthreads();
    float rm = red[r * 8];
    for (int t = 1; t < 8; ++t) rm = fmaxf(rm, red[r * 8 + t]);
    float part = 0.f;
    for (int j = 0; j < 64; ++j) {               // exp once, store back
      float e = expf(Sf[r * SEQ + c0 + j] - rm);
      Sf[r * SEQ + c0 + j] = e;
      part += e;
    }
    __syncthreads();
    red[r * 8 + (threadIdx.x & 7)] = part;
    __syncthreads();
    float rs = 0.f;
    for (int t = 0; t < 8; ++t) rs += red[r * 8 + t];
    float inv = 1.0f / rs;
    int i = it * 16 + r;
    const float* arow = adj  + ((size_t)b_ * SEQ + i) * SEQ + c0;
    const float* drow = dist + ((size_t)b_ * SEQ + i) * SEQ + c0;
    for (int j = 0; j < 64; ++j) {
      float p = Sf[r * SEQ + c0 + j] * inv;                     // LA = 1.0
      float v = p + 0.5f * arow[j] + 0.5f * expf(-drow[j]);     // LG = LD = 0.5
      Abf[r * SEQ + c0 + j] = f2bf(v);
    }
  }
  __syncthreads();

  // ---- out = attn @ v : wave handles one 16-wide d tile (4*16 = 64 = U)
  {
    int d0 = wave * 16;
    v8f c = {};
    const __bf16* vb = vT + ((size_t)bh * UNITS + d0) * SEQ;
    for (int jt = 0; jt < SEQ / 32; ++jt) {
      v16bf a = load_a(&Abf[0] + jt * 32, SEQ, lane);   // ds-fed A fragment
      v16bf b = load_b(vb + jt * 32, SEQ, lane);
      c = wmma_bf(a, b, c);
    }
    int h = bh & 7;
    for (int r = 0; r < 8; ++r) {
      int i = it * 16 + mh + r;
      O[((size_t)b_ * SEQ + i) * INNER + h * UNITS + d0 + n] = f2bf(c[r]);
    }
  }
}

// ---- Stage 3: output projection + bias ----------------------------------
// Block: 4 waves share LDS-staged A (16x512 = 16KB); each wave does 4 col
// tiles -> the whole 256-col output row band per block.
__global__ void out_gemm(const __bf16* __restrict__ X, const __bf16* __restrict__ Wt,
                         const float* __restrict__ bias, float* __restrict__ Y) {
  __shared__ __align__(32) __bf16 As[16 * INNER];        // 16 KB
  int lane = threadIdx.x & 31;
  int wave = threadIdx.x >> 5;
  int rt = blockIdx.x;                   // 512 row tiles

  const __bf16* Arow = X + (size_t)rt * 16 * INNER;
  for (int t = threadIdx.x; t < (16 * INNER) / 8; t += blockDim.x)
    cp16_to_lds(As + t * 8, Arow + t * 8);
  wait_async_lds();
  __syncthreads();

  int n0 = wave * 64;                    // 4 col tiles of 16 per wave
  const __bf16* Bt = Wt + (size_t)n0 * INNER;
  v8f c0 = {}, c1 = {}, c2 = {}, c3 = {};
  for (int kt = 0; kt < INNER / 32; ++kt) {
    v16bf a  = load_a(As + kt * 32, INNER, lane);
    v16bf b0 = load_b(Bt + (size_t) 0 * INNER + kt * 32, INNER, lane);
    v16bf b1 = load_b(Bt + (size_t)16 * INNER + kt * 32, INNER, lane);
    v16bf b2 = load_b(Bt + (size_t)32 * INNER + kt * 32, INNER, lane);
    v16bf b3 = load_b(Bt + (size_t)48 * INNER + kt * 32, INNER, lane);
    c0 = wmma_bf(a, b0, c0);
    c1 = wmma_bf(a, b1, c1);
    c2 = wmma_bf(a, b2, c2);
    c3 = wmma_bf(a, b3, c3);
  }
  int n = lane & 15, mh = (lane >> 4) << 3;
  v8f cc[4] = {c0, c1, c2, c3};
  for (int tcol = 0; tcol < 4; ++tcol) {
    int nn = n0 + tcol * 16 + n;
    float bb = bias[nn];
    for (int r = 0; r < 8; ++r)
      Y[((size_t)rt * 16 + mh + r) * FDIM + nn] = cc[tcol][r] + bb;
  }
}

// ---- host side -----------------------------------------------------------
extern "C" void kernel_launch(void* const* d_in, const int* in_sizes, int n_in,
                              void* d_out, int out_size, void* d_ws, size_t ws_size,
                              hipStream_t stream) {
  const float* node = (const float*)d_in[0];
  const float* adj  = (const float*)d_in[1];
  const float* dist = (const float*)d_in[2];
  const float* Wqkv = (const float*)d_in[6];
  const float* Wout = (const float*)d_in[7];
  const float* bout = (const float*)d_in[8];

  char* ws = (char*)d_ws;
  size_t off = 0;
  auto alloc = [&](size_t bytes) -> void* {
    void* p = ws + off;
    off = (off + bytes + 255) & ~(size_t)255;
    return p;
  };
  const size_t nNode = (size_t)BATCH * SEQ * FDIM;           // 2,097,152
  const size_t nHead = (size_t)BATCH * NHEAD * SEQ * UNITS;  // 4,194,304
  __bf16* nodeBF = (__bf16*)alloc(nNode * 2);
  __bf16* Wqt    = (__bf16*)alloc((size_t)QKVC * FDIM * 2);
  __bf16* Wot    = (__bf16*)alloc((size_t)FDIM * INNER * 2);
  __bf16* qb     = (__bf16*)alloc(nHead * 2);
  __bf16* kb     = (__bf16*)alloc(nHead * 2);
  __bf16* vT     = (__bf16*)alloc(nHead * 2);
  __bf16* Ob     = (__bf16*)alloc((size_t)BATCH * SEQ * INNER * 2);

  cvt_f32_bf16<<<(int)((nNode + 255) / 256), 256, 0, stream>>>(node, nodeBF, (int)nNode);
  transpose_w<<<(FDIM * QKVC + 255) / 256, 256, 0, stream>>>(Wqkv, Wqt, FDIM, QKVC);
  transpose_w<<<(INNER * FDIM + 255) / 256, 256, 0, stream>>>(Wout, Wot, INNER, FDIM);

  qkv_gemm<<<dim3(512, 12), 128, 0, stream>>>(nodeBF, Wqt, qb, kb, vT);
  attn_fused<<<dim3(32, 128), 128, 0, stream>>>(qb, kb, vT, adj, dist, Ob);
  out_gemm<<<dim3(512, 1), 128, 0, stream>>>(Ob, Wot, bout, (float*)d_out);
}
